// Seq2Seq_43327630082249
// MI455X (gfx1250) — matrix operand order repaired and checked by reading, not compiled, not run on hardware
//
#include <hip/hip_runtime.h>

// CDNA5 / gfx1250 seq2seq (B=64, T=1024, D=1, H=256).
// Persistent-workgroup LSTM chains; per-step GEMMs via v_wmma_f32_16x16x32_f16.
// kt-outer / m-inner loop nest: each B tile loaded once per kt (global_load_b128,
// L2-resident) and reused across M rows; A tiles from LDS ping-pong buffers.
// Integer-offset laundering keeps weight loads per-step (no LICM hoist+spill)
// without destroying global address-space provenance.

typedef __attribute__((ext_vector_type(16))) _Float16 v16h;
typedef __attribute__((ext_vector_type(8)))  _Float16 v8h;
typedef __attribute__((ext_vector_type(8)))  float    v8f;

namespace {

constexpr int Bb = 64;     // batch
constexpr int Tt = 1024;   // seq len
constexpr int Hh = 256;    // hidden
constexpr int G4 = 4 * Hh; // 1024 gate lanes

// Fast nonlinearities: v_exp_f32 + v_rcp_f32 (no IEEE div expansion).
__device__ __forceinline__ float sigm(float x) {
  return __builtin_amdgcn_rcpf(1.0f + __expf(-x));
}
__device__ __forceinline__ float tanhe(float x) {
  return 2.0f * __builtin_amdgcn_rcpf(1.0f + __expf(-2.0f * x)) - 1.0f;
}

// Laundered zero offset (SGPR). Adding it to a pointer keeps the GEP rooted at
// the kernel argument (global address space preserved -> global_load saddr form)
// but makes derived addresses opaque per call, so LICM cannot hoist the weight
// loads out of the persistent time loop and spill them to scratch.
__device__ __forceinline__ int opq0() {
  int v = 0;
  asm volatile("" : "+s"(v));
  return v;
}

// A-operand (16x32 f16, MxK) per ISA 7.12.2: lanes 0-15 row M=lane hold K 0..7,16..23;
// lanes 16-31 row M=lane-16 hold K 8..15,24..31  -> two contiguous 16B chunks per lane.
__device__ __forceinline__ v16h load_a16(const _Float16* base, int rowStride, int row,
                                         int kbase, int hl) {
  union U { v16h v; v8h h[2]; } u;
  const _Float16* p = base + row * rowStride + kbase + hl * 8;
  u.h[0] = *(const v8h*)(p);
  u.h[1] = *(const v8h*)(p + 16);
  return u.v;
}

__device__ __forceinline__ v8f wmma16(v16h a, v16h b, v8f c) {
  return __builtin_amdgcn_wmma_f32_16x16x32_f16(false, a, false, b, (short)0, c, false, false);
}

// Pre-swizzle W (f32, row-major [N][K], used as B = W^T) into per-lane-contiguous
// f16 WMMA B-operand tiles: out[((nt*KT+kt)*32+lane)*16+i] = W[nt*16+(lane&15)][kt*32+(i>>3)*16+(lane>>4)*8+(i&7)]
// (B layout matches A: lanes 0-15 hold K 0..7,16..23, lanes 16-31 hold K 8..15,24..31, N = lane%16).
__global__ void k_pack(const float* __restrict__ W, _Float16* __restrict__ out, int Ktot) {
  int idx  = blockIdx.x * blockDim.x + threadIdx.x;
  int i    = idx & 15;
  int lane = (idx >> 4) & 31;
  int rest = idx >> 9;
  int KT   = Ktot >> 5;
  int kt   = rest % KT;
  int nt   = rest / KT;
  int n    = nt * 16 + (lane & 15);
  int k    = kt * 32 + (i >> 3) * 16 + (lane >> 4) * 8 + (i & 7);
  out[idx] = (_Float16)W[n * Ktot + k];
}

// ---------------- Encoder layer 0 (blockIdx.x = direction) ----------------
__global__ __launch_bounds__(512) void k_enc_l0(
    const float* __restrict__ x, const float* __restrict__ wih,
    const float* __restrict__ bias, const _Float16* __restrict__ pw,
    _Float16* __restrict__ y0, float* __restrict__ eh, float* __restrict__ ec) {
  __shared__ __align__(32) _Float16 hbuf[2][Bb * Hh]; // ping-pong f16 h state (2x32KB)
  const int dir = blockIdx.x;
  const int tid = threadIdx.x;
  const int wv = tid >> 5, lane = tid & 31, hl = lane >> 4, ln = lane & 15;
  const int jt = wv, j = jt * 16 + ln;
  const float* wvd = wih + dir * G4;
  const float* bvd = bias + dir * G4;

  for (int i = tid; i < Bb * Hh; i += 512) hbuf[0][i] = (_Float16)0.0f;
  float creg[4][8];
#pragma unroll
  for (int m = 0; m < 4; m++)
#pragma unroll
    for (int r = 0; r < 8; r++) creg[m][r] = 0.0f;

  float xw[4], bg[4];
#pragma unroll
  for (int g = 0; g < 4; g++) { xw[g] = wvd[g * Hh + j]; bg[g] = bvd[g * Hh + j]; }
  __syncthreads();

  for (int s = 0; s < Tt; s++) {
    const int t = dir ? (Tt - 1 - s) : s;
    const _Float16* pwt = pw + (size_t)dir * (Hh * G4) + opq0(); // re-stream from L2 each step
    const _Float16* hrd = hbuf[s & 1];
    _Float16* hwr = hbuf[(s & 1) ^ 1];
#pragma unroll
    for (int mh = 0; mh < 2; mh++) {
      v8f acc[2][4] = {};
#pragma clang loop unroll(disable)
      for (int kt = 0; kt < 8; kt++) {
        v16h bt[4];
#pragma unroll
        for (int g = 0; g < 4; g++) {
          int nt = g * 16 + jt;
          bt[g] = *(const v16h*)(pwt + ((size_t)(nt * 8 + kt) * 32 + lane) * 16);
        }
#pragma unroll
        for (int mi = 0; mi < 2; mi++) {
          v16h a = load_a16(hrd, Hh, (mh * 2 + mi) * 16 + ln, kt * 32, hl);
#pragma unroll
          for (int g = 0; g < 4; g++) acc[mi][g] = wmma16(a, bt[g], acc[mi][g]);
        }
      }
#pragma unroll
      for (int mi = 0; mi < 2; mi++) {
        const int m = mh * 2 + mi;
#pragma unroll
        for (int r = 0; r < 8; r++) {
          int row = m * 16 + hl * 8 + r;
          float xb = x[row * Tt + t];
          float gi = acc[mi][0][r] + xb * xw[0] + bg[0];
          float gf = acc[mi][1][r] + xb * xw[1] + bg[1];
          float gg = acc[mi][2][r] + xb * xw[2] + bg[2];
          float go = acc[mi][3][r] + xb * xw[3] + bg[3];
          float cc = sigm(gf) * creg[m][r] + sigm(gi) * tanhe(gg);
          creg[m][r] = cc;
          float hh = sigm(go) * tanhe(cc);
          hwr[row * Hh + j] = (_Float16)hh;
          y0[((size_t)t * Bb + row) * 512 + dir * Hh + j] = (_Float16)hh;
          if (s == Tt - 1) {
            eh[((size_t)dir * Bb + row) * Hh + j] = hh;
            ec[((size_t)dir * Bb + row) * Hh + j] = cc;
          }
        }
      }
    }
    __syncthreads();
  }
}

// ---------------- Encoder layer 1 (blockIdx.x = direction) ----------------
__global__ __launch_bounds__(512) void k_enc_l1(
    const _Float16* __restrict__ y0, const float* __restrict__ bias,
    const _Float16* __restrict__ pwih, const _Float16* __restrict__ pwhh,
    float* __restrict__ eh, float* __restrict__ ec) {
  __shared__ __align__(32) _Float16 hbuf[2][Bb * Hh];
  const int dir = blockIdx.x;
  const int tid = threadIdx.x;
  const int wv = tid >> 5, lane = tid & 31, hl = lane >> 4, ln = lane & 15;
  const int jt = wv, j = jt * 16 + ln;
  const float* bvd = bias + dir * G4;

  for (int i = tid; i < Bb * Hh; i += 512) hbuf[0][i] = (_Float16)0.0f;
  float creg[4][8];
#pragma unroll
  for (int m = 0; m < 4; m++)
#pragma unroll
    for (int r = 0; r < 8; r++) creg[m][r] = 0.0f;
  float bg[4];
#pragma unroll
  for (int g = 0; g < 4; g++) bg[g] = bvd[g * Hh + j];
  __syncthreads();

  for (int s = 0; s < Tt; s++) {
    const int t = dir ? (Tt - 1 - s) : s;
    const _Float16* pwit = pwih + (size_t)dir * (512 * G4) + opq0();
    const _Float16* pwht = pwhh + (size_t)dir * (Hh * G4) + opq0();
    const _Float16* hrd = hbuf[s & 1];
    _Float16* hwr = hbuf[(s & 1) ^ 1];
    const _Float16* zb = y0 + (size_t)t * Bb * 512;
#pragma unroll
    for (int mh = 0; mh < 2; mh++) {
      v8f acc[2][4] = {};
#pragma clang loop unroll(disable)
      for (int kt = 0; kt < 16; kt++) { // z @ Wih^T (K=512), A from global (L2-hot)
        v16h bt[4];
#pragma unroll
        for (int g = 0; g < 4; g++) {
          int nt = g * 16 + jt;
          bt[g] = *(const v16h*)(pwit + ((size_t)(nt * 16 + kt) * 32 + lane) * 16);
        }
#pragma unroll
        for (int mi = 0; mi < 2; mi++) {
          v16h a = load_a16(zb, 512, (mh * 2 + mi) * 16 + ln, kt * 32, hl);
#pragma unroll
          for (int g = 0; g < 4; g++) acc[mi][g] = wmma16(a, bt[g], acc[mi][g]);
        }
      }
#pragma clang loop unroll(disable)
      for (int kt = 0; kt < 8; kt++) { // h @ Whh^T (K=256), A from LDS
        v16h bt[4];
#pragma unroll
        for (int g = 0; g < 4; g++) {
          int nt = g * 16 + jt;
          bt[g] = *(const v16h*)(pwht + ((size_t)(nt * 8 + kt) * 32 + lane) * 16);
        }
#pragma unroll
        for (int mi = 0; mi < 2; mi++) {
          v16h a = load_a16(hrd, Hh, (mh * 2 + mi) * 16 + ln, kt * 32, hl);
#pragma unroll
          for (int g = 0; g < 4; g++) acc[mi][g] = wmma16(a, bt[g], acc[mi][g]);
        }
      }
#pragma unroll
      for (int mi = 0; mi < 2; mi++) {
        const int m = mh * 2 + mi;
#pragma unroll
        for (int r = 0; r < 8; r++) {
          int row = m * 16 + hl * 8 + r;
          float gi = acc[mi][0][r] + bg[0];
          float gf = acc[mi][1][r] + bg[1];
          float gg = acc[mi][2][r] + bg[2];
          float go = acc[mi][3][r] + bg[3];
          float cc = sigm(gf) * creg[m][r] + sigm(gi) * tanhe(gg);
          creg[m][r] = cc;
          float hh = sigm(go) * tanhe(cc);
          hwr[row * Hh + j] = (_Float16)hh;
          if (s == Tt - 1) {
            eh[((size_t)(2 + dir) * Bb + row) * Hh + j] = hh;
            ec[((size_t)(2 + dir) * Bb + row) * Hh + j] = cc;
          }
        }
      }
    }
    __syncthreads();
  }
}

// ------- Decoder gate constants: Gc[cell][b][n] = eh[cell] @ Whh^T + bias (hoisted) -------
__global__ void k_const(const float* __restrict__ eh,
                        const float* __restrict__ whh0, const float* __restrict__ b0,
                        const float* __restrict__ whh1, const float* __restrict__ b1,
                        float* __restrict__ Gc) {
  int idx = blockIdx.x * blockDim.x + threadIdx.x; // 4*64*1024
  int n = idx & 1023;
  int b = (idx >> 10) & 63;
  int cell = idx >> 16;
  const float* W = (cell < 2 ? whh0 + (size_t)cell * G4 * Hh
                             : whh1 + (size_t)(cell - 2) * G4 * Hh) + (size_t)n * Hh;
  const float* bias = (cell < 2 ? b0 + cell * G4 : b1 + (cell - 2) * G4);
  const float* hv = eh + ((size_t)cell * Bb + b) * Hh;
  float s = 0.0f;
  for (int k = 0; k < Hh; k++) s += hv[k] * W[k];
  Gc[idx] = s + bias[n];
}

// ---------------- Decoder: 1024-step scalar fixed-point iteration ----------------
__global__ __launch_bounds__(512) void k_dec(
    const float* __restrict__ Gc, const float* __restrict__ ec,
    const float* __restrict__ wih0, const _Float16* __restrict__ pw,
    const float* __restrict__ fcW, const float* __restrict__ fcB,
    float* __restrict__ out) {
  extern __shared__ char smem[];
  _Float16* z16 = (_Float16*)smem;                 // 64 x 512 f16 (64KB)
  float* ycur   = (float*)(smem + Bb * 512 * 2);   // 64
  float* ynext  = ycur + Bb;                       // 64
  const int tid = threadIdx.x;
  const int wv = tid >> 5, lane = tid & 31, hl = lane >> 4, ln = lane & 15;
  const int jt = wv, j = jt * 16 + ln;
  if (tid < Bb) ycur[tid] = 0.0f; // x0 = zeros
  __syncthreads();

  for (int t = 0; t < Tt; t++) {
    const float* Gct = Gc + opq0();
    const float* ect = ec + opq0();
    const float* w0t = wih0 + opq0();
    const _Float16* pwb = pw + opq0();
    // ---- stage A: decoder L0 cells (pure elementwise; input is one scalar per row) ----
    if (tid < Bb) ynext[tid] = 0.0f;
#pragma clang loop unroll(disable)
    for (int e = tid; e < 2 * Bb * Hh; e += 512) {
      int jj = e & 255, b = (e >> 8) & 63, d = e >> 14;
      float xv = ycur[b];
      const float* gp = Gct + ((size_t)d * Bb + b) * G4 + jj;
      const float* wp = w0t + d * G4 + jj;
      float gi = gp[0]       + xv * wp[0];
      float gf = gp[Hh]      + xv * wp[Hh];
      float gg = gp[2 * Hh]  + xv * wp[2 * Hh];
      float go = gp[3 * Hh]  + xv * wp[3 * Hh];
      float cc = sigm(gf) * ect[((size_t)d * Bb + b) * Hh + jj] + sigm(gi) * tanhe(gg);
      z16[b * 512 + d * Hh + jj] = (_Float16)(sigm(go) * tanhe(cc));
    }
    __syncthreads();
    // ---- stage B: decoder L1 WMMA GEMM (z @ Wih^T) + epilogue fused with fc dot ----
#pragma clang loop unroll(disable)
    for (int d = 0; d < 2; d++) {
      const _Float16* pwd = pwb + (size_t)d * (512 * G4);
      float fw = fcW[d * Hh + j];
#pragma unroll
      for (int mh = 0; mh < 2; mh++) {
        v8f acc[2][4] = {};
#pragma clang loop unroll(disable)
        for (int kt = 0; kt < 16; kt++) {
          v16h bt[4];
#pragma unroll
          for (int g = 0; g < 4; g++) {
            int nt = g * 16 + jt;
            bt[g] = *(const v16h*)(pwd + ((size_t)(nt * 16 + kt) * 32 + lane) * 16);
          }
#pragma unroll
          for (int mi = 0; mi < 2; mi++) {
            v16h a = load_a16(z16, 512, (mh * 2 + mi) * 16 + ln, kt * 32, hl);
#pragma unroll
            for (int g = 0; g < 4; g++) acc[mi][g] = wmma16(a, bt[g], acc[mi][g]);
          }
        }
#pragma unroll
        for (int mi = 0; mi < 2; mi++) {
          const int m = mh * 2 + mi;
#pragma unroll
          for (int r = 0; r < 8; r++) {
            int row = m * 16 + hl * 8 + r;
            const float* gp = Gct + ((size_t)(2 + d) * Bb + row) * G4 + j;
            float gi = acc[mi][0][r] + gp[0];
            float gf = acc[mi][1][r] + gp[Hh];
            float gg = acc[mi][2][r] + gp[2 * Hh];
            float go = acc[mi][3][r] + gp[3 * Hh];
            float cc = sigm(gf) * ect[((size_t)(2 + d) * Bb + row) * Hh + j] + sigm(gi) * tanhe(gg);
            float v = (sigm(go) * tanhe(cc)) * fw; // fc partial
            v += __shfl_xor(v, 1, 32);
            v += __shfl_xor(v, 2, 32);
            v += __shfl_xor(v, 4, 32);
            v += __shfl_xor(v, 8, 32);
            if (ln == 0) atomicAdd(&ynext[row], v); // ds_add_f32
          }
        }
      }
    }
    __syncthreads();
    // ---- stage C: finish fc, emit output, feed back as next input ----
    if (tid < Bb) {
      float yv = ynext[tid] + fcB[0];
      ycur[tid] = yv;
      out[tid * Tt + t] = yv;
    }
    __syncthreads();
  }
}

} // namespace

extern "C" void kernel_launch(void* const* d_in, const int* in_sizes, int n_in,
                              void* d_out, int out_size, void* d_ws, size_t ws_size,
                              hipStream_t stream) {
  (void)in_sizes; (void)n_in; (void)out_size; (void)ws_size;
  const float* x     = (const float*)d_in[0];
  const float* eWih0 = (const float*)d_in[1];
  const float* eWhh0 = (const float*)d_in[2];
  const float* eB0   = (const float*)d_in[3];
  const float* eWih1 = (const float*)d_in[4];
  const float* eWhh1 = (const float*)d_in[5];
  const float* eB1   = (const float*)d_in[6];
  const float* dWih0 = (const float*)d_in[7];
  const float* dWhh0 = (const float*)d_in[8];
  const float* dB0   = (const float*)d_in[9];
  const float* dWih1 = (const float*)d_in[10];
  const float* dWhh1 = (const float*)d_in[11];
  const float* dB1   = (const float*)d_in[12];
  const float* fcW   = (const float*)d_in[13];
  const float* fcB   = (const float*)d_in[14];
  float* out = (float*)d_out;

  // Workspace carve-up (256B aligned). All weight packs are L2-resident (~6MB);
  // y0 activations (64MB f16) are the only big buffer.
  char* ws = (char*)d_ws;
  size_t off = 0;
  auto take = [&](size_t bytes) { void* p = ws + off; off += (bytes + 255) & ~(size_t)255; return p; };
  _Float16* pwEncWhh0 = (_Float16*)take((size_t)2 * 256 * 1024 * 2);
  _Float16* pwEncWih1 = (_Float16*)take((size_t)2 * 512 * 1024 * 2);
  _Float16* pwEncWhh1 = (_Float16*)take((size_t)2 * 256 * 1024 * 2);
  _Float16* pwDecWih1 = (_Float16*)take((size_t)2 * 512 * 1024 * 2);
  _Float16* y0        = (_Float16*)take((size_t)1024 * 64 * 512 * 2);
  float*    eh        = (float*)take((size_t)4 * 64 * 256 * 4);
  float*    ec        = (float*)take((size_t)4 * 64 * 256 * 4);
  float*    Gc        = (float*)take((size_t)4 * 64 * 1024 * 4);

  struct P { const float* w; _Float16* o; int K; };
  const P packs[8] = {
      {eWhh0,          pwEncWhh0,          256},
      {eWhh0 + 262144, pwEncWhh0 + 262144, 256},
      {eWih1,          pwEncWih1,          512},
      {eWih1 + 524288, pwEncWih1 + 524288, 512},
      {eWhh1,          pwEncWhh1,          256},
      {eWhh1 + 262144, pwEncWhh1 + 262144, 256},
      {dWih1,          pwDecWih1,          512},
      {dWih1 + 524288, pwDecWih1 + 524288, 512},
  };
  for (int p = 0; p < 8; p++) {
    int elems = 1024 * packs[p].K;
    k_pack<<<elems / 256, 256, 0, stream>>>(packs[p].w, packs[p].o, packs[p].K);
  }

  k_enc_l0<<<2, 512, 0, stream>>>(x, eWih0, eB0, pwEncWhh0, y0, eh, ec);
  k_enc_l1<<<2, 512, 0, stream>>>(y0, eB1, pwEncWih1, pwEncWhh1, eh, ec);
  k_const<<<(4 * 64 * 1024) / 256, 256, 0, stream>>>(eh, dWhh0, dB0, dWhh1, dB1, Gc);

  const int decSmem = 64 * 512 * 2 + 2 * 64 * (int)sizeof(float); // 66048B dynamic LDS
  (void)hipFuncSetAttribute((const void*)k_dec,
                            hipFuncAttributeMaxDynamicSharedMemorySize, decSmem);
  k_dec<<<1, 512, decSmem, stream>>>(Gc, ec, dWih0, pwDecWih1, fcW, fcB, out);
}
